// EigenMLP_BN_53377853554931
// MI455X (gfx1250) — matrix operand order repaired and checked by reading, not compiled, not run on hardware
//
#include <hip/hip_runtime.h>
#include <hip/hip_bf16.h>

#define N_NODES 50000
#define N_EDGES 800000
#define N_GRAPHS 64
#define PERIOD 10
#define HID 300
#define HID2 600
#define NLAYER 5
#define EPS_BN 1e-5f

typedef __attribute__((ext_vector_type(16))) __bf16 v16bf;
typedef __attribute__((ext_vector_type(8)))  float  v8f;

// ---------------------------------------------------------------------------
// Generic zero-fill
// ---------------------------------------------------------------------------
__global__ void zero_f32(float* __restrict__ p, size_t n) {
    size_t i = blockIdx.x * (size_t)blockDim.x + threadIdx.x;
    if (i < n) p[i] = 0.f;
}

// ---------------------------------------------------------------------------
// Convert f32 activations [M x K] -> zero-padded bf16 [M x Kpad]
// ---------------------------------------------------------------------------
__global__ void pack_act(const float* __restrict__ X, __bf16* __restrict__ Y,
                         int M, int K, int Kpad) {
    size_t idx = blockIdx.x * (size_t)blockDim.x + threadIdx.x;
    size_t total = (size_t)M * Kpad;
    if (idx >= total) return;
    int    k = (int)(idx % Kpad);
    size_t m = idx / Kpad;
    Y[idx] = (__bf16)((k < K) ? X[m * (size_t)K + k] : 0.f);
}

// ---------------------------------------------------------------------------
// Pack f32 weight [K x N] into bf16 WMMA B-fragments.
// Tile count is padded to an EVEN number of 16-wide tiles (zero fill) so the
// GEMM can block 2 N-tiles per wave.
//   Bp[ ((nt*KT + kt)*32 + lane)*16 + i ] = W[kt*32 + (lane>=16)*8 + (i&7) + (i>>3)*16][nt*16 + lane%16]
// (16-bit B 32x16: lanes 0-15 hold K=0-7,16-23, lanes 16-31 hold K=8-15,24-31; N = lane%16)
// ---------------------------------------------------------------------------
__global__ void pack_weight(const float* __restrict__ W, __bf16* __restrict__ Bp,
                            int K, int N, int Kpad) {
    int    KT    = Kpad >> 5;
    int    ntile = ((N + 31) >> 5) * 2;            // even # of 16-wide tiles
    size_t total = (size_t)ntile * KT * 32 * 16;
    size_t idx = blockIdx.x * (size_t)blockDim.x + threadIdx.x;
    if (idx >= total) return;
    int    i    = (int)(idx & 15);
    int    lane = (int)((idx >> 4) & 31);
    size_t rest = idx >> 9;
    int    kt   = (int)(rest % KT);
    int    nt   = (int)(rest / KT);
    int n = nt * 16 + (lane & 15);
    int k = kt * 32 + (lane >> 4) * 8 + (i & 7) + (i >> 3) * 16;
    float v = (n < N && k < K) ? W[(size_t)k * N + n] : 0.f;
    Bp[idx] = (__bf16)v;
}

// ---------------------------------------------------------------------------
// C[M x N] = A(bf16, M x Kpad row-major) @ Bp(packed fragments) + bias.
// Each wave32 computes a 32x32 output block = 2x2 grid of 16x16 WMMA tiles:
// 4 f32 accumulators, A fragments reused across the 2 N-tiles and B fragments
// across the 2 M-tiles -> 8 b128 loads per 4 v_wmma_f32_16x16x32_bf16.
// A fragment (ISA 16-bit A 16x32): lanes 0-15: halves 0-7 = K 0-7, 8-15 = K 16-23;
// lanes 16-31: halves 0-7 = K 8-15, 8-15 = K 24-31 -> two contiguous 16B loads.
// C/D: row = tileBase + 8*(lane>=16) + r, col = tileBase + lane%16.
// Out-of-range rows (M not a multiple of 32) are clamped for loads (rows are
// independent in GEMM) and masked at the store.
// ---------------------------------------------------------------------------
__global__ void wmma_gemm_bias(const __bf16* __restrict__ A,
                               const __bf16* __restrict__ Bp,
                               const float* __restrict__ bias,
                               float* __restrict__ C,
                               int M, int N, int Kpad) {
    const int lane = threadIdx.x & 31;
    const int wave = (int)((blockIdx.x * (size_t)blockDim.x + threadIdx.x) >> 5);
    const int nt2 = (N + 31) >> 5;                 // 32-col supertiles
    const int mt2 = (M + 31) >> 5;                 // 32-row supertiles
    if (wave >= mt2 * nt2) return;                 // wave-uniform exit
    const int mt = wave / nt2;
    const int nt = wave % nt2;
    const int KT = Kpad >> 5;
    const int hs = lane >> 4;                      // 0: lanes 0-15, 1: lanes 16-31
    const int lm = lane & 15;

    int row0 = mt * 32 + lm;      if (row0 >= M) row0 = M - 1;
    int row1 = mt * 32 + 16 + lm; if (row1 >= M) row1 = M - 1;
    const __bf16* a0p = A + (size_t)row0 * Kpad + hs * 8;
    const __bf16* a1p = A + (size_t)row1 * Kpad + hs * 8;
    const __bf16* b0p = Bp + ((size_t)(nt * 2)     * KT * 32 + lane) * 16;
    const __bf16* b1p = Bp + ((size_t)(nt * 2 + 1) * KT * 32 + lane) * 16;

    v8f acc00 = {}, acc01 = {}, acc10 = {}, acc11 = {};
    for (int kt = 0; kt < KT; ++kt) {
        v16bf a0, a1, b0, b1;
        uint4* p;
        p = reinterpret_cast<uint4*>(&a0);
        p[0] = *reinterpret_cast<const uint4*>(a0p + (size_t)kt * 32);
        p[1] = *reinterpret_cast<const uint4*>(a0p + (size_t)kt * 32 + 16);
        p = reinterpret_cast<uint4*>(&a1);
        p[0] = *reinterpret_cast<const uint4*>(a1p + (size_t)kt * 32);
        p[1] = *reinterpret_cast<const uint4*>(a1p + (size_t)kt * 32 + 16);
        p = reinterpret_cast<uint4*>(&b0);
        const uint4* bs0 = reinterpret_cast<const uint4*>(b0p + (size_t)kt * 512);
        p[0] = bs0[0];
        p[1] = bs0[1];
        p = reinterpret_cast<uint4*>(&b1);
        const uint4* bs1 = reinterpret_cast<const uint4*>(b1p + (size_t)kt * 512);
        p[0] = bs1[0];
        p[1] = bs1[1];
        __builtin_prefetch(a0p + (size_t)(kt + 1) * 32, 0, 1);  // global_prefetch_b8
        __builtin_prefetch(a1p + (size_t)(kt + 1) * 32, 0, 1);
        acc00 = __builtin_amdgcn_wmma_f32_16x16x32_bf16(false, a0, false, b0, (short)0, acc00, false, false);
        acc01 = __builtin_amdgcn_wmma_f32_16x16x32_bf16(false, a0, false, b1, (short)0, acc01, false, false);
        acc10 = __builtin_amdgcn_wmma_f32_16x16x32_bf16(false, a1, false, b0, (short)0, acc10, false, false);
        acc11 = __builtin_amdgcn_wmma_f32_16x16x32_bf16(false, a1, false, b1, (short)0, acc11, false, false);
    }

    const int col0 = nt * 32 + lm;
    const int col1 = col0 + 16;
    const int r00  = mt * 32 + hs * 8;       // rows for acc0x
    const int r10  = r00 + 16;               // rows for acc1x
    if (col0 < N) {
        const float bv = bias[col0];
#pragma unroll
        for (int r = 0; r < 8; ++r) {
            int row = r00 + r;
            if (row < M) C[(size_t)row * N + col0] = acc00[r] + bv;
        }
#pragma unroll
        for (int r = 0; r < 8; ++r) {
            int row = r10 + r;
            if (row < M) C[(size_t)row * N + col0] = acc10[r] + bv;
        }
    }
    if (col1 < N) {
        const float bv = bias[col1];
#pragma unroll
        for (int r = 0; r < 8; ++r) {
            int row = r00 + r;
            if (row < M) C[(size_t)row * N + col1] = acc01[r] + bv;
        }
#pragma unroll
        for (int r = 0; r < 8; ++r) {
            int row = r10 + r;
            if (row < M) C[(size_t)row * N + col1] = acc11[r] + bv;
        }
    }
}

// ---------------------------------------------------------------------------
// agg[dst] += ew * h[src]  — one thread per (edge, 4-feature group)
// ---------------------------------------------------------------------------
__global__ void edge_scatter(const int* __restrict__ ei, const float* __restrict__ ew,
                             const float* __restrict__ h, float* __restrict__ agg) {
    const int G = HID / 4;                        // 75 float4 groups
    size_t idx = blockIdx.x * (size_t)blockDim.x + threadIdx.x;
    size_t total = (size_t)N_EDGES * G;
    if (idx >= total) return;
    int g = (int)(idx % G);
    int e = (int)(idx / G);
    int s = ei[e];
    int d = ei[N_EDGES + e];
    float w = ew[e];
    const float4 hv = *reinterpret_cast<const float4*>(h + (size_t)s * HID + g * 4);
    float* out = agg + (size_t)d * HID + g * 4;
    unsafeAtomicAdd(out + 0, w * hv.x);
    unsafeAtomicAdd(out + 1, w * hv.y);
    unsafeAtomicAdd(out + 2, w * hv.z);
    unsafeAtomicAdd(out + 3, w * hv.w);
}

// ---------------------------------------------------------------------------
// BN column stats: coalesced partial sums + atomics, then finalize mean/rstd
// ---------------------------------------------------------------------------
__global__ void bn_stats_partial(const float* __restrict__ T, int M, int N,
                                 float* __restrict__ sums, float* __restrict__ sumsq) {
    int col = blockIdx.x * blockDim.x + threadIdx.x;
    if (col >= N) return;
    float s = 0.f, ss = 0.f;
    for (int r = blockIdx.y; r < M; r += gridDim.y) {
        float v = T[(size_t)r * N + col];
        s += v;
        ss += v * v;
    }
    unsafeAtomicAdd(&sums[col], s);
    unsafeAtomicAdd(&sumsq[col], ss);
}

__global__ void bn_finalize(const float* __restrict__ sums, const float* __restrict__ sumsq,
                            int M, int N, float* __restrict__ mean, float* __restrict__ rstd) {
    int col = blockIdx.x * blockDim.x + threadIdx.x;
    if (col >= N) return;
    float m = sums[col] / (float)M;
    float v = sumsq[col] / (float)M - m * m;
    mean[col] = m;
    rstd[col] = rsqrtf(v + EPS_BN);
}

// BN apply + relu -> padded bf16 (feeds next WMMA GEMM directly)
__global__ void bn_apply_bf16(const float* __restrict__ T, const float* __restrict__ mean,
                              const float* __restrict__ rstd, const float* __restrict__ g,
                              const float* __restrict__ b, int M, int N, int Kpad,
                              __bf16* __restrict__ out) {
    size_t idx = blockIdx.x * (size_t)blockDim.x + threadIdx.x;
    size_t total = (size_t)M * Kpad;
    if (idx >= total) return;
    int    k = (int)(idx % Kpad);
    size_t m = idx / Kpad;
    float v = 0.f;
    if (k < N) {
        v = g[k] * (T[m * (size_t)N + k] - mean[k]) * rstd[k] + b[k];
        v = fmaxf(v, 0.f);
    }
    out[idx] = (__bf16)v;
}

// BN apply (optional relu) -> f32 h
__global__ void bn_apply_f32(const float* __restrict__ T, const float* __restrict__ mean,
                             const float* __restrict__ rstd, const float* __restrict__ g,
                             const float* __restrict__ b, int M, int N,
                             float* __restrict__ out, int relu) {
    size_t idx = blockIdx.x * (size_t)blockDim.x + threadIdx.x;
    size_t total = (size_t)M * N;
    if (idx >= total) return;
    int k = (int)(idx % N);
    float v = g[k] * (T[idx] - mean[k]) * rstd[k] + b[k];
    if (relu) v = fmaxf(v, 0.f);
    out[idx] = v;
}

// ---------------------------------------------------------------------------
// xpool[batch[n]] += h[n]
// ---------------------------------------------------------------------------
__global__ void pool_kernel(const float* __restrict__ h, const int* __restrict__ batch,
                            float* __restrict__ xpool) {
    const int G = HID / 4;
    size_t idx = blockIdx.x * (size_t)blockDim.x + threadIdx.x;
    size_t total = (size_t)N_NODES * G;
    if (idx >= total) return;
    int g = (int)(idx % G);
    int n = (int)(idx / G);
    int bg = batch[n];
    const float4 hv = *reinterpret_cast<const float4*>(h + (size_t)n * HID + g * 4);
    float* out = xpool + (size_t)bg * HID + g * 4;
    unsafeAtomicAdd(out + 0, hv.x);
    unsafeAtomicAdd(out + 1, hv.y);
    unsafeAtomicAdd(out + 2, hv.z);
    unsafeAtomicAdd(out + 3, hv.w);
}

// ---------------------------------------------------------------------------
extern "C" void kernel_launch(void* const* d_in, const int* in_sizes, int n_in,
                              void* d_out, int out_size, void* d_ws, size_t ws_size,
                              hipStream_t stream) {
    (void)in_sizes; (void)n_in; (void)out_size; (void)ws_size;

    const int*   batch = (const int*)  d_in[0];
    const float* x     = (const float*)d_in[1];
    const int*   eidx  = (const int*)  d_in[2];
    const float* eattr = (const float*)d_in[3];
    const float* linW  = (const float*)d_in[4];
    const float* linb  = (const float*)d_in[5];
    const float* W1s   = (const float*)d_in[6];
    const float* b1s   = (const float*)d_in[7];
    const float* g1s   = (const float*)d_in[8];
    const float* be1s  = (const float*)d_in[9];
    const float* W2s   = (const float*)d_in[10];
    const float* b2s   = (const float*)d_in[11];
    const float* gos   = (const float*)d_in[12];
    const float* bos   = (const float*)d_in[13];

    float* h     = (float*)d_out;                               // [N_NODES x HID]
    float* xpool = (float*)d_out + (size_t)N_NODES * HID;       // [N_GRAPHS x HID]

    // --- carve workspace (256B aligned slabs) ---
    char*  ws  = (char*)d_ws;
    size_t off = 0;
    auto carve = [&](size_t bytes) -> char* {
        char* p = ws + off;
        off += (bytes + 255) & ~(size_t)255;
        return p;
    };
    __bf16* actA  = (__bf16*)carve((size_t)N_NODES * 320 * 2);  // bf16 GEMM1/input A
    __bf16* actT  = (__bf16*)carve((size_t)N_NODES * 640 * 2);  // bf16 GEMM2 A
    float*  agg   = (float*) carve((size_t)N_NODES * HID * 4);  // agg / GEMM2 raw out
    float*  t1    = (float*) carve((size_t)N_NODES * HID2 * 4); // GEMM1 raw out
    __bf16* pW    = (__bf16*)carve(1 << 19);                    // packed weights (reused)
    float*  sums  = (float*) carve(640 * 4);
    float*  sumsq = (float*) carve(640 * 4);
    float*  mean  = (float*) carve(640 * 4);
    float*  rstd  = (float*) carve(640 * 4);
    (void)carve(64 * 32);                                        // guard slab for clamped rows

    const int TB = 256;
    auto blocks = [&](size_t n) -> unsigned { return (unsigned)((n + TB - 1) / TB); };

    auto launch_packw = [&](const float* W, int K, int N, int Kpad) {
        int ntile = ((N + 31) / 32) * 2;
        size_t total = (size_t)ntile * (Kpad / 32) * 512;
        pack_weight<<<blocks(total), TB, 0, stream>>>(W, pW, K, N, Kpad);
    };
    auto launch_gemm = [&](const __bf16* A, const float* bias, float* C, int N, int Kpad) {
        size_t waves = (size_t)((N_NODES + 31) / 32) * ((N + 31) / 32);
        wmma_gemm_bias<<<blocks(waves * 32), TB, 0, stream>>>(A, pW, bias, C, N_NODES, N, Kpad);
    };
    auto bn_stats = [&](const float* T, int N) {
        zero_f32<<<blocks(N), TB, 0, stream>>>(sums, (size_t)N);
        zero_f32<<<blocks(N), TB, 0, stream>>>(sumsq, (size_t)N);
        dim3 g((N + TB - 1) / TB, 512);
        bn_stats_partial<<<g, TB, 0, stream>>>(T, N_NODES, N, sums, sumsq);
        bn_finalize<<<blocks(N), TB, 0, stream>>>(sums, sumsq, N_NODES, N, mean, rstd);
    };

    // ---- input linear: h = x @ linW + linb  (K = 20 padded to 32) ----
    pack_act<<<blocks((size_t)N_NODES * 32), TB, 0, stream>>>(x, actA, N_NODES, 2 * PERIOD, 32);
    launch_packw(linW, 2 * PERIOD, HID, 32);
    launch_gemm(actA, linb, h, HID, 32);

    // ---- 5 message-passing layers ----
    for (int i = 0; i < NLAYER; ++i) {
        // agg = segment_sum(ew * h[src], dst)
        zero_f32<<<blocks((size_t)N_NODES * HID), TB, 0, stream>>>(agg, (size_t)N_NODES * HID);
        edge_scatter<<<blocks((size_t)N_EDGES * (HID / 4)), TB, 0, stream>>>(eidx, eattr, h, agg);

        // t = relu(BN(agg @ W1 + b1))
        pack_act<<<blocks((size_t)N_NODES * 320), TB, 0, stream>>>(agg, actA, N_NODES, HID, 320);
        launch_packw(W1s + (size_t)i * HID * HID2, HID, HID2, 320);
        launch_gemm(actA, b1s + (size_t)i * HID2, t1, HID2, 320);
        bn_stats(t1, HID2);
        bn_apply_bf16<<<blocks((size_t)N_NODES * 640), TB, 0, stream>>>(
            t1, mean, rstd, g1s + (size_t)i * HID2, be1s + (size_t)i * HID2,
            N_NODES, HID2, 640, actT);

        // h = BN(t @ W2 + b2) [+ relu if not last]
        launch_packw(W2s + (size_t)i * HID2 * HID, HID2, HID, 640);
        launch_gemm(actT, b2s + (size_t)i * HID, agg, HID, 640);
        bn_stats(agg, HID);
        bn_apply_f32<<<blocks((size_t)N_NODES * HID), TB, 0, stream>>>(
            agg, mean, rstd, gos + (size_t)i * HID, bos + (size_t)i * HID,
            N_NODES, HID, h, (i != NLAYER - 1) ? 1 : 0);
    }

    // ---- graph pooling ----
    zero_f32<<<blocks((size_t)N_GRAPHS * HID), TB, 0, stream>>>(xpool, (size_t)N_GRAPHS * HID);
    pool_kernel<<<blocks((size_t)N_NODES * (HID / 4)), TB, 0, stream>>>(h, batch, xpool);
}
